// MultiHeadSelfAttention_25632364822786
// MI455X (gfx1250) — matrix operand order repaired and checked by reading, not compiled, not run on hardware
//
#include <hip/hip_runtime.h>
#include <hip/hip_bf16.h>
#include <math.h>

// ---------------------------------------------------------------------------
// MI455X (gfx1250) multi-head self-attention
//   bf16 WMMA + Tensor Data Mover (async LDS staging, double buffered)
//   B=1, S=4096, D=1024, H=16, HD=64
// ---------------------------------------------------------------------------

typedef __attribute__((ext_vector_type(16))) __bf16 v16bf;
typedef __attribute__((ext_vector_type(8)))  float  v8f;
typedef __attribute__((ext_vector_type(4))) unsigned int u32x4;
typedef __attribute__((ext_vector_type(8)))  int i32x8;
typedef __attribute__((ext_vector_type(4)))  int i32x4;

union Frag16 {              // 16 bf16 = 32 bytes = one WMMA A/B operand (8 VGPRs)
    uint4  u[2];
    v16bf  v;
};

constexpr int kS  = 4096;
constexpr int kD  = 1024;
constexpr int kH  = 16;
constexpr int kHD = 64;

// ===========================================================================
// TDM: DMA one 2D bf16 tile (tileRows x tileK, row stride strideElems) from
// global into LDS at byte offset ldsOff. Issued once by the executing wave.
// D# per CDNA5 ISA §8.3-8.5: count=1, data_size=2B, type=2, no pad/iterate.
// 6-arg builtin form (clang-23 / therock-10.0 headers).
// ===========================================================================
__device__ __forceinline__ void tdm_load_2d(const __bf16* gsrc, unsigned ldsOff,
                                            int tileK, int tileRows,
                                            int strideElems) {
    unsigned long long ga = (unsigned long long)(uintptr_t)gsrc;
    u32x4 g0;
    g0.x = 1u;                                        // count=1 (valid user D#)
    g0.y = ldsOff;                                    // lds_addr (bytes)
    g0.z = (unsigned)(ga & 0xFFFFFFFFu);              // global_addr[31:0]
    g0.w = (unsigned)((ga >> 32) & 0x01FFFFFFu) | (2u << 30);  // addr[56:32]|type=2

    const unsigned td0 = 1u << 20;                    // oversized tensor dims:
    const unsigned td1 = 1u << 20;                    //   tile never clamps OOB
    i32x8 g1;
    g1[0] = (int)(1u << 16);                          // wg_mask=0, data_size=1(2B)
    g1[1] = (int)((td0 & 0xFFFFu) << 16);             // tensor_dim0[15:0]
    g1[2] = (int)((td0 >> 16) | ((td1 & 0xFFFFu) << 16));
    g1[3] = (int)((td1 >> 16) | ((unsigned)tileK << 16));   // tile_dim0
    g1[4] = tileRows;                                 // tile_dim1 (tile_dim2=0)
    g1[5] = strideElems;                              // tensor_dim0_stride lo32
    g1[6] = 0;
    g1[7] = 0;
    i32x4 g2 = {0, 0, 0, 0};
    i32x4 g3 = {0, 0, 0, 0};
    i32x8 g4 = {0, 0, 0, 0, 0, 0, 0, 0};
    __builtin_amdgcn_tensor_load_to_lds(g0, g1, g2, g3, g4, 0);
}

// ===========================================================================
// Elementwise fp32 -> bf16 (activations). n multiple of 8.
// ===========================================================================
__global__ __launch_bounds__(256)
void convert_f32_bf16(const float* __restrict__ src, __bf16* __restrict__ dst,
                      int n) {
    int i = (blockIdx.x * 256 + threadIdx.x) * 8;
    if (i >= n) return;
    float4 f0 = *(const float4*)(src + i);
    float4 f1 = *(const float4*)(src + i + 4);
    union { __bf16 h[8]; uint4 u; } pk;
    pk.h[0] = (__bf16)f0.x; pk.h[1] = (__bf16)f0.y;
    pk.h[2] = (__bf16)f0.z; pk.h[3] = (__bf16)f0.w;
    pk.h[4] = (__bf16)f1.x; pk.h[5] = (__bf16)f1.y;
    pk.h[6] = (__bf16)f1.z; pk.h[7] = (__bf16)f1.w;
    *(uint4*)(dst + i) = pk.u;
}

// ===========================================================================
// Weight transpose + convert: Wt[n][k] = (bf16)W[k][n], 1024x1024.
// Tile 64x64 through LDS; grid (16,16).
// ===========================================================================
__global__ __launch_bounds__(256)
void transpose_f32_bf16(const float* __restrict__ W, __bf16* __restrict__ Wt) {
    __shared__ __bf16 tile[64 * 65];
    const int t  = threadIdx.x;
    const int rB = blockIdx.y * 64, cB = blockIdx.x * 64;
    #pragma unroll
    for (int i = 0; i < 16; ++i) {
        int idx = t + i * 256;              // 0..4095
        int r = idx >> 6, c = idx & 63;     // coalesced over c
        tile[c * 65 + r] = (__bf16)W[(size_t)(rB + r) * kD + cB + c];
    }
    __syncthreads();
    #pragma unroll
    for (int i = 0; i < 8; ++i) {
        int idx = t + i * 256;              // 0..2047
        int n = idx >> 5, rp = (idx & 31) * 2;
        union { __bf16 h[2]; unsigned u; } pk;
        pk.h[0] = tile[n * 65 + rp];
        pk.h[1] = tile[n * 65 + rp + 1];
        *(unsigned*)&Wt[(size_t)(cB + n) * kD + rB + rp] = pk.u;
    }
}

// ===========================================================================
// GEMM: C[4096x1024] = A[4096x1024](bf16) * Bt[n][k](bf16, pre-transposed)
//       + bias, f32 accum. Macro tile 128x128, 8 waves of 32x64, k-step 64.
// Panels staged by TDM (tensor_load_to_lds), double buffered.
//   mode 0: bf16 row-major [S][D]   mode 1: bf16 transposed [D][S]
//   mode 2: fp32 row-major [S][D]
// ===========================================================================
__global__ __launch_bounds__(256)
void gemm_bf16_tdm(const __bf16* __restrict__ A, const __bf16* __restrict__ Bt,
                   const float* __restrict__ bias, void* __restrict__ Cout,
                   int mode) {
    // double buffer: [buf][A(128x64) | B(128x64)]  = 64 KB total
    __shared__ __bf16 lds[2][2 * 128 * 64];

    const int t    = threadIdx.x;
    const int lane = t & 31;
    const int wave = t >> 5;
    const int l16  = lane & 15;
    const int lh   = lane >> 4;

    const int rowBase = blockIdx.y * 128;
    const int colBase = blockIdx.x * 128;
    const int wr = (wave & 3) * 32;          // wave row offset (2 subtiles)
    const int wc = (wave >> 2) * 64;         // wave col offset (4 subtiles)

    const unsigned ldsA[2] = { (unsigned)(uintptr_t)(void*)&lds[0][0],
                               (unsigned)(uintptr_t)(void*)&lds[1][0] };
    const unsigned ldsB[2] = { (unsigned)(uintptr_t)(void*)&lds[0][128 * 64],
                               (unsigned)(uintptr_t)(void*)&lds[1][128 * 64] };

    v8f acc[8];
    #pragma unroll
    for (int i = 0; i < 8; ++i) acc[i] = (v8f){};

    // ---- preload tile 0 (wave 0 drives the TDM) ----
    if (wave == 0) {
        tdm_load_2d(&A [(size_t)rowBase * kD + 0], ldsA[0], 64, 128, kD);
        tdm_load_2d(&Bt[(size_t)colBase * kD + 0], ldsB[0], 64, 128, kD);
        __builtin_amdgcn_s_wait_tensorcnt(0);
    }
    __syncthreads();

    int cur = 0;
    for (int k0 = 0; k0 < kD; k0 += 64) {
        // issue next tile's DMA into the other buffer (overlaps compute)
        const bool more = (k0 + 64) < kD;
        if (wave == 0 && more) {
            tdm_load_2d(&A [(size_t)rowBase * kD + k0 + 64], ldsA[cur ^ 1], 64, 128, kD);
            tdm_load_2d(&Bt[(size_t)colBase * kD + k0 + 64], ldsB[cur ^ 1], 64, 128, kD);
        }

        const __bf16* Ab = &lds[cur][0];
        const __bf16* Bb = &lds[cur][128 * 64];
        #pragma unroll
        for (int kc = 0; kc < 64; kc += 32) {
            Frag16 a[2];
            #pragma unroll
            for (int ms = 0; ms < 2; ++ms) {
                const uint4* ap = (const uint4*)&Ab[(wr + ms * 16 + l16) * 64 + kc + lh * 8];
                a[ms].u[0] = ap[0];
                a[ms].u[1] = ap[2];
            }
            Frag16 b[4];
            #pragma unroll
            for (int ns = 0; ns < 4; ++ns) {
                const uint4* bp = (const uint4*)&Bb[(wc + ns * 16 + l16) * 64 + kc + lh * 8];
                b[ns].u[0] = bp[0];
                b[ns].u[1] = bp[2];
            }
            #pragma unroll
            for (int ms = 0; ms < 2; ++ms)
                #pragma unroll
                for (int ns = 0; ns < 4; ++ns)
                    acc[ms * 4 + ns] = __builtin_amdgcn_wmma_f32_16x16x32_bf16(
                        false, a[ms].v, false, b[ns].v, (short)0,
                        acc[ms * 4 + ns], false, false);
        }

        __syncthreads();                              // everyone done with cur
        if (wave == 0 && more) __builtin_amdgcn_s_wait_tensorcnt(0);
        __syncthreads();                              // next buffer published
        cur ^= 1;
    }

    // ---- epilogue ----
    #pragma unroll
    for (int ns = 0; ns < 4; ++ns) {
        const int col = colBase + wc + ns * 16 + l16;
        const float bb = bias[col];
        #pragma unroll
        for (int ms = 0; ms < 2; ++ms) {
            #pragma unroll
            for (int i = 0; i < 8; ++i) {
                int row = rowBase + wr + ms * 16 + lh * 8 + i;
                float c = acc[ms * 4 + ns][i] + bb;
                if (mode == 0)
                    ((__bf16*)Cout)[(size_t)row * kD + col] = (__bf16)c;
                else if (mode == 1)
                    ((__bf16*)Cout)[(size_t)col * kS + row] = (__bf16)c;
                else
                    ((float*)Cout)[(size_t)row * kD + col] = c;
            }
        }
    }
}

// ===========================================================================
// Flash attention (causal), one wave per 16-query tile of one head.
// Block = 256 threads = 8 waves = 128 query rows; grid = (S/128, H).
// qp, kp: bf16 [S][D].  vpT: bf16 [D][S].  ctx out: bf16 [S][D].
// ===========================================================================
__global__ __launch_bounds__(256)
void attn_flash_wmma(const __bf16* __restrict__ qp, const __bf16* __restrict__ kp,
                     const __bf16* __restrict__ vpT, __bf16* __restrict__ ctx) {
    __shared__ __bf16 Pl[8][16 * 32];   // per-wave probability tile

    const int t    = threadIdx.x;
    const int lane = t & 31;
    const int wave = t >> 5;
    const int l16  = lane & 15;
    const int lh   = lane >> 4;

    const int h  = blockIdx.y;
    const int q0 = blockIdx.x * 128 + wave * 16;

    Frag16 a0, a1;
    {
        const uint4* qpp =
            (const uint4*)&qp[(size_t)(q0 + l16) * kD + h * kHD + lh * 8];
        a0.u[0] = qpp[0]; a0.u[1] = qpp[2];
        a1.u[0] = qpp[4]; a1.u[1] = qpp[6];
    }

    v8f c0 = {}, c1 = {}, c2 = {}, c3 = {};
    float m_[8], l_[8];
    #pragma unroll
    for (int i = 0; i < 8; ++i) { m_[i] = -INFINITY; l_[i] = 0.0f; }

    const int nK = (q0 + 47) >> 5;
    const float scale = 0.125f;         // 1/sqrt(64)

    for (int j = 0; j < nK; ++j) {
        const int kb = j * 32;

        Frag16 bk00, bk01, bk10, bk11;
        {
            const uint4* p0 =
                (const uint4*)&kp[(size_t)(kb + l16) * kD + h * kHD + lh * 8];
            bk00.u[0] = p0[0]; bk00.u[1] = p0[2];
            bk01.u[0] = p0[4]; bk01.u[1] = p0[6];
            const uint4* p1 =
                (const uint4*)&kp[(size_t)(kb + 16 + l16) * kD + h * kHD + lh * 8];
            bk10.u[0] = p1[0]; bk10.u[1] = p1[2];
            bk11.u[0] = p1[4]; bk11.u[1] = p1[6];
        }

        v8f s0 = {}, s1 = {};
        s0 = __builtin_amdgcn_wmma_f32_16x16x32_bf16(false, a0.v, false, bk00.v,
                                                     (short)0, s0, false, false);
        s0 = __builtin_amdgcn_wmma_f32_16x16x32_bf16(false, a1.v, false, bk01.v,
                                                     (short)0, s0, false, false);
        s1 = __builtin_amdgcn_wmma_f32_16x16x32_bf16(false, a0.v, false, bk10.v,
                                                     (short)0, s1, false, false);
        s1 = __builtin_amdgcn_wmma_f32_16x16x32_bf16(false, a1.v, false, bk11.v,
                                                     (short)0, s1, false, false);

        float p0v[8], p1v[8];
        #pragma unroll
        for (int i = 0; i < 8; ++i) {
            int qrow = q0 + lh * 8 + i;
            p0v[i] = (kb + l16      <= qrow) ? s0[i] * scale : -INFINITY;
            p1v[i] = (kb + 16 + l16 <= qrow) ? s1[i] * scale : -INFINITY;
        }

        #pragma unroll
        for (int i = 0; i < 8; ++i) {
            float tmax = fmaxf(p0v[i], p1v[i]);
            #pragma unroll
            for (int mks = 1; mks < 16; mks <<= 1)
                tmax = fmaxf(tmax, __shfl_xor(tmax, mks, 32));
            float mo = m_[i];
            float mn = fmaxf(mo, tmax);
            float alpha = (mo == -INFINITY) ? 0.0f : __expf(mo - mn);
            float e0 = __expf(p0v[i] - mn);
            float e1 = __expf(p1v[i] - mn);
            p0v[i] = e0; p1v[i] = e1;
            float sum = e0 + e1;
            #pragma unroll
            for (int mks = 1; mks < 16; mks <<= 1)
                sum += __shfl_xor(sum, mks, 32);
            l_[i] = l_[i] * alpha + sum;
            m_[i] = mn;
            c0[i] *= alpha; c1[i] *= alpha; c2[i] *= alpha; c3[i] *= alpha;
        }

        __bf16* pl = Pl[wave];
        #pragma unroll
        for (int i = 0; i < 8; ++i) {
            pl[(lh * 8 + i) * 32 + l16]      = (__bf16)p0v[i];
            pl[(lh * 8 + i) * 32 + 16 + l16] = (__bf16)p1v[i];
        }
        asm volatile("s_wait_dscnt 0" ::: "memory");

        Frag16 pa;
        const uint4* pp = (const uint4*)&pl[l16 * 32 + lh * 8];
        pa.u[0] = pp[0]; pa.u[1] = pp[2];

        #pragma unroll
        for (int nt = 0; nt < 4; ++nt) {
            Frag16 bv_;
            const uint4* vp_ = (const uint4*)
                &vpT[(size_t)(h * kHD + nt * 16 + l16) * kS + kb + lh * 8];
            bv_.u[0] = vp_[0]; bv_.u[1] = vp_[2];
            if (nt == 0)
                c0 = __builtin_amdgcn_wmma_f32_16x16x32_bf16(false, pa.v, false, bv_.v,
                                                             (short)0, c0, false, false);
            else if (nt == 1)
                c1 = __builtin_amdgcn_wmma_f32_16x16x32_bf16(false, pa.v, false, bv_.v,
                                                             (short)0, c1, false, false);
            else if (nt == 2)
                c2 = __builtin_amdgcn_wmma_f32_16x16x32_bf16(false, pa.v, false, bv_.v,
                                                             (short)0, c2, false, false);
            else
                c3 = __builtin_amdgcn_wmma_f32_16x16x32_bf16(false, pa.v, false, bv_.v,
                                                             (short)0, c3, false, false);
        }
        asm volatile("s_wait_dscnt 0" ::: "memory");
    }

    #pragma unroll
    for (int i = 0; i < 8; ++i) {
        float inv = 1.0f / l_[i];
        int row = q0 + lh * 8 + i;
        __bf16* crow = &ctx[(size_t)row * kD + h * kHD];
        crow[l16]      = (__bf16)(c0[i] * inv);
        crow[16 + l16] = (__bf16)(c1[i] * inv);
        crow[32 + l16] = (__bf16)(c2[i] * inv);
        crow[48 + l16] = (__bf16)(c3[i] * inv);
    }
}

// ===========================================================================
// Launch
// ===========================================================================
extern "C" void kernel_launch(void* const* d_in, const int* in_sizes, int n_in,
                              void* d_out, int out_size, void* d_ws, size_t ws_size,
                              hipStream_t stream) {
    const float* q_in = (const float*)d_in[0];
    const float* k_in = (const float*)d_in[1];
    const float* v_in = (const float*)d_in[2];
    // d_in[3] = causal mask (implemented analytically)
    const float* Wq = (const float*)d_in[4];
    const float* bq = (const float*)d_in[5];
    const float* Wk = (const float*)d_in[6];
    const float* bk = (const float*)d_in[7];
    const float* Wv = (const float*)d_in[8];
    const float* bv = (const float*)d_in[9];
    const float* Wo = (const float*)d_in[10];
    const float* bo = (const float*)d_in[11];
    float* out = (float*)d_out;

    char* ws = (char*)d_ws;
    const size_t MB = 1024 * 1024;
    __bf16* xq   = (__bf16*)(ws);             // converted inputs, 8 MB each
    __bf16* xk   = (__bf16*)(ws + 8  * MB);
    __bf16* xv   = (__bf16*)(ws + 16 * MB);
    __bf16* Wqt  = (__bf16*)(ws + 24 * MB);   // transposed bf16 weights, 2 MB each
    __bf16* Wkt  = (__bf16*)(ws + 26 * MB);
    __bf16* Wvt  = (__bf16*)(ws + 28 * MB);
    __bf16* Wot  = (__bf16*)(ws + 30 * MB);
    __bf16* qp   = (__bf16*)(ws + 32 * MB);   // projected Q/K, 8 MB each
    __bf16* kp   = (__bf16*)(ws + 40 * MB);
    __bf16* vpT  = (__bf16*)(ws + 48 * MB);   // projected V, transposed
    __bf16* ctxb = (__bf16*)(ws + 56 * MB);   // attention context (bf16)

    dim3 blk(256);
    const int nElem = kS * kD;                // 4M
    convert_f32_bf16<<<nElem / (8 * 256), blk, 0, stream>>>(q_in, xq, nElem);
    convert_f32_bf16<<<nElem / (8 * 256), blk, 0, stream>>>(k_in, xk, nElem);
    convert_f32_bf16<<<nElem / (8 * 256), blk, 0, stream>>>(v_in, xv, nElem);

    dim3 gT(16, 16);
    transpose_f32_bf16<<<gT, blk, 0, stream>>>(Wq, Wqt);
    transpose_f32_bf16<<<gT, blk, 0, stream>>>(Wk, Wkt);
    transpose_f32_bf16<<<gT, blk, 0, stream>>>(Wv, Wvt);
    transpose_f32_bf16<<<gT, blk, 0, stream>>>(Wo, Wot);

    dim3 gGemm(kD / 128, kS / 128);           // 8 x 32
    gemm_bf16_tdm<<<gGemm, blk, 0, stream>>>(xq, Wqt, bq, qp,  0);
    gemm_bf16_tdm<<<gGemm, blk, 0, stream>>>(xk, Wkt, bk, kp,  0);
    gemm_bf16_tdm<<<gGemm, blk, 0, stream>>>(xv, Wvt, bv, vpT, 1);

    dim3 gAttn(kS / 128, kH);                 // 32 x 16
    attn_flash_wmma<<<gAttn, blk, 0, stream>>>(qp, kp, vpT, ctxb);

    gemm_bf16_tdm<<<gGemm, blk, 0, stream>>>(ctxb, Wot, bo, out, 2);
}
